// PointerGeneratorNetwork_56495999811933
// MI455X (gfx1250) — matrix operand order repaired and checked by reading, not compiled, not run on hardware
//
#include <hip/hip_runtime.h>
#include <cstdint>
#include <cstddef>

typedef __attribute__((ext_vector_type(16))) _Float16 v16h;
typedef __attribute__((ext_vector_type(8)))  float    v8f;

#define Bq     32
#define TINq   256
#define TTGTq  100
#define Vq     50000
#define Eq     128
#define Hq     160
#define Dq     196
#define Aq     224
#define BNq    56
#define NVq    49998      /* V - END */
#define NVP    50000      /* NVq padded to 16 (=3125*16) */
#define H2q    320
#define G4Hq   640
#define G4Dq   784
#define DPq    224        /* D padded to mult of 32 */
#define HCPq   544        /* D+2H=516 padded */
#define KZq    64         /* BN=56 padded */
#define GWAVES 4

__device__ __forceinline__ float sigf(float x) { return 1.f / (1.f + __expf(-x)); }

// ---------------- WMMA fragment loaders (wave32, 16x16x32 f16) ----------------
// A fragment: lanes 0-15 hold M=lane, K=k0..k0+15;
//             lanes 16-31 hold M=lane-16, K=k0+16..k0+31.
__device__ __forceinline__ v16h load_fragA(const _Float16* src, int ld, int mBase, int k0, int lane) {
  int r  = lane & 15;
  int hk = (lane >> 4) << 4;
  const _Float16* p = src + (size_t)(mBase + r) * ld + k0 + hk;
  v16h f;
#pragma unroll
  for (int i = 0; i < 8; ++i) f[i] = p[i];
#pragma unroll
  for (int i = 0; i < 8; ++i) f[i + 8] = p[8 + i];
  return f;
}
// B fragment: lanes 0-15 hold N=lane, K=k0..k0+15; lanes 16-31 hold N=lane-16, K=k0+16..k0+31.
// W is stored [N][K] row-major, so each lane reads 16 contiguous halves.
__device__ __forceinline__ v16h load_fragB(const _Float16* src, int ld, int nBase, int k0, int lane) {
  int r  = lane & 15;
  int hk = (lane >> 4) << 4;
  const _Float16* p = src + (size_t)(nBase + r) * ld + k0 + hk;
  v16h f;
#pragma unroll
  for (int i = 0; i < 16; ++i) f[i] = p[i];
  return f;
}

// ---------------- generic WMMA GEMM: C = [C +] A(MxK) @ W(NxK)^T + bias ----------------
// requires K % 32 == 0, M % 16 == 0, N % 16 == 0 (callers pad). Branch-free inner loop.
__global__ __launch_bounds__(32 * GWAVES)
void gemm_wmma_kernel(const _Float16* __restrict__ A, int lda,
                      const _Float16* __restrict__ W, int ldw,
                      const float* __restrict__ bias,
                      float* __restrict__ C, int ldc,
                      int N, int K, int accum, int act) {
  int lane  = threadIdx.x & 31;
  int wave  = threadIdx.x >> 5;
  int nBase = (blockIdx.x * GWAVES + wave) * 16;
  int mBase = blockIdx.y * 16;
  if (nBase >= N) return;
  v8f acc = {0.f,0.f,0.f,0.f,0.f,0.f,0.f,0.f};
  for (int k0 = 0; k0 < K; k0 += 32) {
    v16h a = load_fragA(A, lda, mBase, k0, lane);
    v16h b = load_fragB(W, ldw, nBase, k0, lane);
    acc = __builtin_amdgcn_wmma_f32_16x16x32_f16(false, a, false, b, (short)0, acc, false, false);
  }
  int col = nBase + (lane & 15);
  int rb  = mBase + ((lane >> 4) << 3);
  float bv = bias ? bias[col] : 0.f;
#pragma unroll
  for (int j = 0; j < 8; ++j) {
    float v = acc[j] + bv;
    if (act == 1) v = tanhf(v);
    size_t o = (size_t)(rb + j) * ldc + col;
    if (accum) C[o] += v; else C[o] = v;
  }
}

// ---------------- weight convert, row+col pad, to f16 ----------------
__global__ void convert_pad_f16(const float* __restrict__ src, _Float16* __restrict__ dst,
                                int N, int K, int Np, int Kp) {
  int i = blockIdx.x * blockDim.x + threadIdx.x;
  if (i >= Np * Kp) return;
  int n = i / Kp, k = i - n * Kp;
  dst[i] = (n < N && k < K) ? (_Float16)src[(size_t)n * K + k] : (_Float16)0.f;
}

__global__ void pad_f32_kernel(const float* __restrict__ src, float* __restrict__ dst,
                               int N, int Np, float fill) {
  int i = blockIdx.x * blockDim.x + threadIdx.x;
  if (i >= Np) return;
  dst[i] = (i < N) ? src[i] : fill;
}

// ---------------- encoder input embedding gather: x16[t*B+b][E] ----------------
__global__ void embed_input_kernel(const int* __restrict__ ids, const _Float16* __restrict__ emb16,
                                   _Float16* __restrict__ x16) {
  int i = blockIdx.x * blockDim.x + threadIdx.x;
  if (i >= Bq * TINq * Eq) return;
  int e  = i & (Eq - 1);
  int bt = i >> 7;
  int b  = bt >> 8;          // bt = b*256 + t
  int t  = bt & (TINq - 1);
  int id = ids[b * TINq + t];
  if (id >= Vq) id = 3;      // UNK
  x16[((size_t)t * Bq + b) * Eq + e] = emb16[(size_t)id * Eq + e];
}

// ---------------- persistent bidirectional encoder LSTM (1 block per direction) ----------------
__global__ void encoder_lstm_kernel(
    const float* __restrict__ xWi_f, const float* __restrict__ xWi_b,
    const _Float16* __restrict__ Whf, const _Float16* __restrict__ Whb,
    const float* __restrict__ bif, const float* __restrict__ bhf,
    const float* __restrict__ bib, const float* __restrict__ bhb,
    float* __restrict__ enc, _Float16* __restrict__ enc16,
    _Float16* __restrict__ hT16, _Float16* __restrict__ cT16) {
  const int dir = blockIdx.x;
  const float*    xWi = dir ? xWi_b : xWi_f;
  const _Float16* Wh  = dir ? Whb   : Whf;
  const float*    bi  = dir ? bib   : bif;
  const float*    bh  = dir ? bhb   : bhf;
  extern __shared__ char smem[];
  float*    hS  = (float*)smem;                 // [32][160]
  float*    cS  = hS + Bq * Hq;                 // [32][160]
  _Float16* h16 = (_Float16*)(cS + Bq * Hq);    // [32][160]
  float*    gS  = (float*)(h16 + Bq * Hq);      // [32][640]
  int tid  = threadIdx.x;
  int lane = tid & 31;
  int wave = tid >> 5;
  int nw   = blockDim.x >> 5;
  for (int u = tid; u < Bq * Hq; u += blockDim.x) { hS[u] = 0.f; cS[u] = 0.f; h16[u] = (_Float16)0.f; }
  __syncthreads();
  for (int t = 0; t < TINq; ++t) {
    int ts = dir ? (TINq - 1 - t) : t;
    const float* xg = xWi + (size_t)ts * Bq * G4Hq;
    // g = h16(32x160) @ Wh(640x160)^T  -> 2 x 40 tiles of 16x16, K=160 (5 chunks)
    for (int tile = wave; tile < 80; tile += nw) {
      int mBase = (tile & 1) * 16;
      int nBase = (tile >> 1) * 16;
      v8f acc = {0.f,0.f,0.f,0.f,0.f,0.f,0.f,0.f};
#pragma unroll
      for (int kc = 0; kc < 5; ++kc) {
        v16h a = load_fragA(h16, Hq, mBase, kc * 32, lane);
        v16h b = load_fragB(Wh, Hq, nBase, kc * 32, lane);
        acc = __builtin_amdgcn_wmma_f32_16x16x32_f16(false, a, false, b, (short)0, acc, false, false);
      }
      int col = nBase + (lane & 15);
      int rb  = mBase + ((lane >> 4) << 3);
      float bb = bi[col] + bh[col];
#pragma unroll
      for (int j = 0; j < 8; ++j) {
        int row = rb + j;
        gS[row * G4Hq + col] = acc[j] + xg[(size_t)row * G4Hq + col] + bb;
      }
    }
    __syncthreads();
    for (int u = tid; u < Bq * Hq; u += blockDim.x) {
      int b = u / Hq, j = u - b * Hq;
      const float* g = gS + b * G4Hq;
      float ig = sigf(g[j]);
      float fg = sigf(g[Hq + j]);
      float gg = tanhf(g[2 * Hq + j]);
      float og = sigf(g[3 * Hq + j]);
      float c = fg * cS[u] + ig * gg;
      float h = og * tanhf(c);
      cS[u] = c; hS[u] = h; h16[u] = (_Float16)h;
      size_t eo = ((size_t)b * TINq + ts) * H2q + dir * Hq + j;
      enc[eo] = h; enc16[eo] = (_Float16)h;
    }
    __syncthreads();
  }
  for (int u = tid; u < Bq * Hq; u += blockDim.x) {
    int b = u / Hq, j = u - b * Hq;
    hT16[b * H2q + dir * Hq + j] = (_Float16)hS[u];
    cT16[b * H2q + dir * Hq + j] = (_Float16)cS[u];
  }
}

// ---------------- decoder-side small kernels ----------------
__global__ void init_state_kernel(float* cov, float* nll, float* cl, int* cur, _Float16* dh16) {
  int i = blockIdx.x * blockDim.x + threadIdx.x;
  if (i < Bq * TINq) cov[i] = 0.f;
  if (i < Bq) { nll[i] = 0.f; cl[i] = 0.f; cur[i] = 1; /* START */ }
  if (i < Bq * DPq) dh16[i] = (_Float16)0.f;
}

__global__ void pack_h16_kernel(const float* __restrict__ h, _Float16* __restrict__ h16) {
  int i = blockIdx.x * blockDim.x + threadIdx.x;
  if (i >= Bq * DPq) return;
  h16[i] = (_Float16)h[i];   // dec_h is ld=224 padded, pads are zero
}

__global__ void embed_dec_kernel(const int* __restrict__ cur, const _Float16* __restrict__ emb16,
                                 _Float16* __restrict__ e16) {
  int i = blockIdx.x * blockDim.x + threadIdx.x;
  if (i >= Bq * Eq) return;
  int b = i >> 7, e = i & (Eq - 1);
  e16[i] = emb16[(size_t)cur[b] * Eq + e];
}

__global__ void dec_lstm_elem_kernel(const float* __restrict__ gates,
                                     float* __restrict__ h, float* __restrict__ c,
                                     _Float16* __restrict__ h16) {
  int i = blockIdx.x * blockDim.x + threadIdx.x;
  if (i >= Bq * Dq) return;
  int b = i / Dq, j = i - b * Dq;
  const float* g = gates + (size_t)b * G4Dq;
  float ig = sigf(g[j]);
  float fg = sigf(g[Dq + j]);
  float gg = tanhf(g[2 * Dq + j]);
  float og = sigf(g[3 * Dq + j]);
  float cc = fg * c[b * DPq + j] + ig * gg;
  float hh = og * tanhf(cc);
  c[b * DPq + j] = cc; h[b * DPq + j] = hh;
  h16[b * DPq + j] = (_Float16)hh;   // pad cols 196..223 stay zero (set at init)
}

__global__ void attn_score_kernel(const float* __restrict__ enc_attn, const float* __restrict__ hWda,
                                  const float* __restrict__ cov, const float* __restrict__ wca,
                                  const float* __restrict__ vat, const int* __restrict__ in_ids,
                                  float* __restrict__ sbuf) {
  int i = blockIdx.x * blockDim.x + threadIdx.x;
  if (i >= Bq * TINq) return;
  int b = i >> 8;
  if (in_ids[i] == 0) { sbuf[i] = -1e30f; return; }  // PAD mask
  float cv = cov[i];
  const float* ea = enc_attn + (size_t)i * Aq;
  const float* hd = hWda + (size_t)b * Aq;
  float acc = 0.f;
  for (int a = 0; a < Aq; ++a)
    acc += vat[a] * tanhf(ea[a] + hd[a] + cv * wca[a]);
  sbuf[i] = acc;
}

// one block (256 threads) per batch row: softmax + ctx + coverage + copy_p + hc pack
__global__ void softmax_ctx_kernel(const float* __restrict__ sbuf, const float* __restrict__ enc,
                                   const int* __restrict__ in_ids, const int* __restrict__ tgt, int tstep,
                                   float* __restrict__ cov, float* __restrict__ cl,
                                   float* __restrict__ copy_p, float* __restrict__ ctx,
                                   const float* __restrict__ dec_h, _Float16* __restrict__ hc16) {
  __shared__ float red[TINq];
  __shared__ float attn_s[TINq];
  __shared__ float ctx_s[H2q];
  int b = blockIdx.x, t = threadIdx.x;
  float sv = sbuf[b * TINq + t];
  red[t] = sv; __syncthreads();
  for (int s = TINq / 2; s > 0; s >>= 1) { if (t < s) red[t] = fmaxf(red[t], red[t + s]); __syncthreads(); }
  float m = red[0]; __syncthreads();
  float e = __expf(sv - m);
  red[t] = e; __syncthreads();
  for (int s = TINq / 2; s > 0; s >>= 1) { if (t < s) red[t] += red[t + s]; __syncthreads(); }
  float Z = red[0]; __syncthreads();
  float attn = e / Z;
  attn_s[t] = attn;
  int nt = tgt[b * TTGTq + tstep];
  float cp = (in_ids[b * TINq + t] == nt) ? attn : 0.f;
  float co = cov[b * TINq + t];
  float mn = fminf(attn, co);
  cov[b * TINq + t] = co + attn;
  red[t] = cp; __syncthreads();
  for (int s = TINq / 2; s > 0; s >>= 1) { if (t < s) red[t] += red[t + s]; __syncthreads(); }
  if (t == 0) copy_p[b] = red[0];
  __syncthreads();
  red[t] = mn; __syncthreads();
  for (int s = TINq / 2; s > 0; s >>= 1) { if (t < s) red[t] += red[t + s]; __syncthreads(); }
  if (t == 0) cl[b] += red[0];
  __syncthreads();
  for (int col = t; col < H2q; col += TINq) {
    float acc = 0.f;
    for (int tt = 0; tt < TINq; ++tt)
      acc += attn_s[tt] * enc[((size_t)b * TINq + tt) * H2q + col];
    ctx_s[col] = acc;
    ctx[b * H2q + col] = acc;
  }
  __syncthreads();
  for (int j = t; j < HCPq; j += TINq) {
    float v = (j < Dq) ? dec_h[b * DPq + j] : (j < Dq + H2q) ? ctx_s[j - Dq] : 0.f;
    hc16[(size_t)b * HCPq + j] = (_Float16)v;
  }
}

// pack z1 (tanh'd, [32][64] padded, pads already 0) to f16, compute p_gen, reset accumulators
__global__ void z1pack_pgen_kernel(const float* __restrict__ z1, _Float16* __restrict__ z1p,
                                   const float* __restrict__ ctx, const float* __restrict__ dec_h,
                                   const int* __restrict__ cur, const float* __restrict__ emb,
                                   const float* __restrict__ Wcs, const float* __restrict__ bcs,
                                   const float* __restrict__ Wds, const float* __restrict__ Wes,
                                   float* __restrict__ pgen, float* __restrict__ sumexp,
                                   float* __restrict__ zsel) {
  int i = threadIdx.x;
  for (int u = i; u < Bq * KZq; u += blockDim.x)
    z1p[u] = (_Float16)z1[u];
  if (i < Bq) {
    float acc = bcs[0];
    const float* cx = ctx + i * H2q;
    for (int j = 0; j < H2q; ++j) acc += cx[j] * Wcs[j];
    const float* hh = dec_h + i * DPq;
    for (int j = 0; j < Dq; ++j) acc += hh[j] * Wds[j];
    const float* ee = emb + (size_t)cur[i] * Eq;
    for (int j = 0; j < Eq; ++j) acc += ee[j] * Wes[j];
    pgen[i] = sigf(acc);
    sumexp[i] = 0.f;
    zsel[i] = 0.f;
  }
}

// fused vocab GEMM + exp + row-sum + selected-logit snipe. 8 waves/block, 1 N-tile/wave.
// W2p padded to NVP rows (zeros); b2p pad = -1e30 so padded cols add exp(-1e30)=0. Branch-free.
__global__ __launch_bounds__(256)
void vocab_kernel(const _Float16* __restrict__ z1p, const _Float16* __restrict__ W2p,
                  const float* __restrict__ b2p, const int* __restrict__ tgt, int tstep,
                  float* __restrict__ sumexp, float* __restrict__ zsel) {
  __shared__ float part[Bq];
  if (threadIdx.x < Bq) part[threadIdx.x] = 0.f;
  __syncthreads();
  int lane = threadIdx.x & 31;
  int wave = threadIdx.x >> 5;
  int nBase = (blockIdx.x * 8 + wave) * 16;
  if (nBase < NVP) {
    int col = nBase + (lane & 15);
    float bv = b2p[col];
#pragma unroll
    for (int mT = 0; mT < 2; ++mT) {
      v8f acc = {0.f,0.f,0.f,0.f,0.f,0.f,0.f,0.f};
#pragma unroll
      for (int kc = 0; kc < 2; ++kc) {
        v16h a = load_fragA(z1p, KZq, mT * 16, kc * 32, lane);
        v16h b = load_fragB(W2p, KZq, nBase, kc * 32, lane);
        acc = __builtin_amdgcn_wmma_f32_16x16x32_f16(false, a, false, b, (short)0, acc, false, false);
      }
      int rb = mT * 16 + ((lane >> 4) << 3);
#pragma unroll
      for (int j = 0; j < 8; ++j) {
        int row = rb + j;
        float z = acc[j] + bv;
        atomicAdd(&part[row], __expf(z));
        int nt = tgt[row * TTGTq + tstep];
        int idx = nt - 2; if (idx < 0) idx = 0; if (idx > NVq - 1) idx = NVq - 1;
        if (col == idx) zsel[row] = z;
      }
    }
  }
  __syncthreads();
  if (threadIdx.x < Bq) atomicAdd(&sumexp[threadIdx.x], part[threadIdx.x]);
}

__global__ void finalize_step_kernel(const int* __restrict__ tgt, int tstep,
                                     const float* __restrict__ zsel, const float* __restrict__ sumexp,
                                     const float* __restrict__ pgen, const float* __restrict__ copy_p,
                                     float* __restrict__ nll, int* __restrict__ cur) {
  int b = threadIdx.x;
  if (b >= Bq) return;
  int nt = tgt[b * TTGTq + tstep];
  float gen_p = 0.f;
  if (nt >= 2 && nt < Vq) gen_p = __expf(zsel[b]) / sumexp[b];
  float pg = pgen[b];
  float p = pg * gen_p + (1.f - pg) * copy_p[b];
  nll[b] -= __logf(p + 1e-9f);
  cur[b] = (nt >= Vq) ? 3 : nt;
}

__global__ void final_out_kernel(const float* __restrict__ nll, const float* __restrict__ cl,
                                 float* __restrict__ out) {
  if (threadIdx.x == 0) {
    float a = 0.f, b = 0.f;
    for (int i = 0; i < Bq; ++i) { a += nll[i]; b += cl[i]; }
    out[0] = a; out[1] = b; out[2] = a + 0.75f * b;
  }
}

// ================================ host side ================================
static inline void gemm_launch(hipStream_t s, const _Float16* A, int lda, const _Float16* W, int ldw,
                               const float* bias, float* C, int ldc, int M, int N, int K,
                               int accum, int act) {
  dim3 g((((N + 15) / 16) + GWAVES - 1) / GWAVES, M / 16);
  gemm_wmma_kernel<<<g, dim3(32 * GWAVES), 0, s>>>(A, lda, W, ldw, bias, C, ldc, N, K, accum, act);
}
static inline void conv_launch(hipStream_t s, const float* src, _Float16* dst,
                               int N, int K, int Np, int Kp) {
  int n = Np * Kp;
  convert_pad_f16<<<(n + 255) / 256, 256, 0, s>>>(src, dst, N, K, Np, Kp);
}

extern "C" void kernel_launch(void* const* d_in, const int* in_sizes, int n_in,
                              void* d_out, int out_size, void* d_ws, size_t ws_size,
                              hipStream_t stream) {
  (void)in_sizes; (void)n_in; (void)out_size; (void)ws_size;
  const int*   in_ids = (const int*)  d_in[0];
  const int*   tgt    = (const int*)  d_in[1];
  const float* emb    = (const float*)d_in[3];
  const float* Wi_f = (const float*)d_in[4];
  const float* Wh_f = (const float*)d_in[5];
  const float* bi_f = (const float*)d_in[6];
  const float* bh_f = (const float*)d_in[7];
  const float* Wi_b = (const float*)d_in[8];
  const float* Wh_b = (const float*)d_in[9];
  const float* bi_b = (const float*)d_in[10];
  const float* bh_b = (const float*)d_in[11];
  const float* Weh  = (const float*)d_in[12];
  const float* beh  = (const float*)d_in[13];
  const float* Wec  = (const float*)d_in[14];
  const float* bec  = (const float*)d_in[15];
  const float* Wi_d = (const float*)d_in[16];
  const float* Wh_d = (const float*)d_in[17];
  const float* bi_d = (const float*)d_in[18];
  const float* bh_d = (const float*)d_in[19];
  const float* Wea  = (const float*)d_in[20];
  const float* bea  = (const float*)d_in[21];
  const float* Wda  = (const float*)d_in[22];
  const float* Wca  = (const float*)d_in[23];
  const float* vat  = (const float*)d_in[24];
  const float* W1   = (const float*)d_in[25];
  const float* b1   = (const float*)d_in[26];
  const float* W2   = (const float*)d_in[27];
  const float* b2   = (const float*)d_in[28];
  const float* Wcs  = (const float*)d_in[29];
  const float* bcs  = (const float*)d_in[30];
  const float* Wds  = (const float*)d_in[31];
  const float* Wes  = (const float*)d_in[32];

  size_t off = 0;
  auto alloc = [&](size_t bytes) -> void* {
    void* p = (void*)((char*)d_ws + off);
    off += (bytes + 255) & ~(size_t)255;
    return p;
  };
  _Float16* emb16  = (_Float16*)alloc((size_t)Vq * Eq * 2);
  _Float16* Wif16  = (_Float16*)alloc((size_t)G4Hq * Eq * 2);
  _Float16* Whf16  = (_Float16*)alloc((size_t)G4Hq * Hq * 2);
  _Float16* Wib16  = (_Float16*)alloc((size_t)G4Hq * Eq * 2);
  _Float16* Whb16  = (_Float16*)alloc((size_t)G4Hq * Hq * 2);
  _Float16* Weh16  = (_Float16*)alloc((size_t)DPq * H2q * 2);   // 196->224 rows
  _Float16* Wec16  = (_Float16*)alloc((size_t)DPq * H2q * 2);
  _Float16* Wid16  = (_Float16*)alloc((size_t)G4Dq * Eq * 2);
  _Float16* Whd16  = (_Float16*)alloc((size_t)G4Dq * DPq * 2);
  _Float16* Wea16  = (_Float16*)alloc((size_t)Aq * H2q * 2);
  _Float16* Wda16  = (_Float16*)alloc((size_t)Aq * DPq * 2);
  _Float16* W116   = (_Float16*)alloc((size_t)KZq * HCPq * 2);  // 56->64 rows
  _Float16* W216   = (_Float16*)alloc((size_t)NVP * KZq * 2);   // 49998->50000 rows
  _Float16* x16    = (_Float16*)alloc((size_t)TINq * Bq * Eq * 2);
  float*    xWif   = (float*)   alloc((size_t)TINq * Bq * G4Hq * 4);
  float*    xWib   = (float*)   alloc((size_t)TINq * Bq * G4Hq * 4);
  float*    enc    = (float*)   alloc((size_t)Bq * TINq * H2q * 4);
  _Float16* enc16  = (_Float16*)alloc((size_t)Bq * TINq * H2q * 2);
  float*    encat  = (float*)   alloc((size_t)Bq * TINq * Aq * 4);
  _Float16* hT16   = (_Float16*)alloc((size_t)Bq * H2q * 2);
  _Float16* cT16   = (_Float16*)alloc((size_t)Bq * H2q * 2);
  float*    behp   = (float*)   alloc(DPq * 4);
  float*    becp   = (float*)   alloc(DPq * 4);
  float*    b1p    = (float*)   alloc(KZq * 4);
  float*    b2p    = (float*)   alloc((size_t)NVP * 4);
  float*    dec_h  = (float*)   alloc((size_t)Bq * DPq * 4);    // ld=224, pads zero
  float*    dec_c  = (float*)   alloc((size_t)Bq * DPq * 4);
  _Float16* dh16   = (_Float16*)alloc((size_t)Bq * DPq * 2);
  _Float16* e16    = (_Float16*)alloc((size_t)Bq * Eq * 2);
  float*    gates  = (float*)   alloc((size_t)Bq * G4Dq * 4);
  float*    hWda   = (float*)   alloc((size_t)Bq * Aq * 4);
  float*    sbuf   = (float*)   alloc((size_t)Bq * TINq * 4);
  float*    cov    = (float*)   alloc((size_t)Bq * TINq * 4);
  float*    ctx    = (float*)   alloc((size_t)Bq * H2q * 4);
  _Float16* hc16   = (_Float16*)alloc((size_t)Bq * HCPq * 2);
  float*    z1     = (float*)   alloc((size_t)Bq * KZq * 4);    // [32][64], pads tanh(0)=0
  _Float16* z1p    = (_Float16*)alloc((size_t)Bq * KZq * 2);
  float*    pgen   = (float*)   alloc(Bq * 4);
  float*    copyp  = (float*)   alloc(Bq * 4);
  float*    zsel   = (float*)   alloc(Bq * 4);
  float*    sume   = (float*)   alloc(Bq * 4);
  float*    nll    = (float*)   alloc(Bq * 4);
  float*    cl     = (float*)   alloc(Bq * 4);
  int*      cur    = (int*)     alloc(Bq * 4);

  // ---- weight conversion (f32 -> f16, N and K padded) ----
  conv_launch(stream, emb,  emb16,  Vq,   Eq,        Vq,   Eq);
  conv_launch(stream, Wi_f, Wif16,  G4Hq, Eq,        G4Hq, Eq);
  conv_launch(stream, Wh_f, Whf16,  G4Hq, Hq,        G4Hq, Hq);
  conv_launch(stream, Wi_b, Wib16,  G4Hq, Eq,        G4Hq, Eq);
  conv_launch(stream, Wh_b, Whb16,  G4Hq, Hq,        G4Hq, Hq);
  conv_launch(stream, Weh,  Weh16,  Dq,   H2q,       DPq,  H2q);
  conv_launch(stream, Wec,  Wec16,  Dq,   H2q,       DPq,  H2q);
  conv_launch(stream, Wi_d, Wid16,  G4Dq, Eq,        G4Dq, Eq);
  conv_launch(stream, Wh_d, Whd16,  G4Dq, Dq,        G4Dq, DPq);
  conv_launch(stream, Wea,  Wea16,  Aq,   H2q,       Aq,   H2q);
  conv_launch(stream, Wda,  Wda16,  Aq,   Dq,        Aq,   DPq);
  conv_launch(stream, W1,   W116,   BNq,  Dq + H2q,  KZq,  HCPq);
  conv_launch(stream, W2,   W216,   NVq,  BNq,       NVP,  KZq);
  pad_f32_kernel<<<1, DPq, 0, stream>>>(beh, behp, Dq, DPq, 0.f);
  pad_f32_kernel<<<1, DPq, 0, stream>>>(bec, becp, Dq, DPq, 0.f);
  pad_f32_kernel<<<1, KZq, 0, stream>>>(b1, b1p, BNq, KZq, 0.f);
  pad_f32_kernel<<<(NVP + 255) / 256, 256, 0, stream>>>(b2, b2p, NVq, NVP, -1e30f);

  // ---- encoder: embed, hoisted input GEMMs, persistent recurrence ----
  {
    int n = Bq * TINq * Eq;
    embed_input_kernel<<<(n + 255) / 256, 256, 0, stream>>>(in_ids, emb16, x16);
  }
  gemm_launch(stream, x16, Eq, Wif16, Eq, nullptr, xWif, G4Hq, TINq * Bq, G4Hq, Eq, 0, 0);
  gemm_launch(stream, x16, Eq, Wib16, Eq, nullptr, xWib, G4Hq, TINq * Bq, G4Hq, Eq, 0, 0);
  size_t enc_smem = (size_t)Bq * Hq * 4 * 2 + (size_t)Bq * Hq * 2 + (size_t)Bq * G4Hq * 4; // 133120 B
  (void)hipFuncSetAttribute(reinterpret_cast<const void*>(encoder_lstm_kernel),
                            hipFuncAttributeMaxDynamicSharedMemorySize, (int)enc_smem);
  encoder_lstm_kernel<<<dim3(2), dim3(512), enc_smem, stream>>>(
      xWif, xWib, Whf16, Whb16, bi_f, bh_f, bi_b, bh_b, enc, enc16, hT16, cT16);

  // ---- attention precompute + decoder init state ----
  gemm_launch(stream, enc16, H2q, Wea16, H2q, bea, encat, Aq, Bq * TINq, Aq, H2q, 0, 0);
  gemm_launch(stream, hT16, H2q, Weh16, H2q, behp, dec_h, DPq, Bq, DPq, H2q, 0, 0);
  gemm_launch(stream, cT16, H2q, Wec16, H2q, becp, dec_c, DPq, Bq, DPq, H2q, 0, 0);
  init_state_kernel<<<32, 256, 0, stream>>>(cov, nll, cl, cur, dh16);
  pack_h16_kernel<<<(Bq * DPq + 255) / 256, 256, 0, stream>>>(dec_h, dh16);

  // ---- decoder loop (serial over 100 target steps) ----
  for (int tstep = 0; tstep < TTGTq; ++tstep) {
    embed_dec_kernel<<<(Bq * Eq + 255) / 256, 256, 0, stream>>>(cur, emb16, e16);
    gemm_launch(stream, e16,  Eq,  Wid16, Eq,  bi_d, gates, G4Dq, Bq, G4Dq, Eq,  0, 0);
    gemm_launch(stream, dh16, DPq, Whd16, DPq, bh_d, gates, G4Dq, Bq, G4Dq, DPq, 1, 0);
    dec_lstm_elem_kernel<<<(Bq * Dq + 255) / 256, 256, 0, stream>>>(gates, dec_h, dec_c, dh16);
    gemm_launch(stream, dh16, DPq, Wda16, DPq, nullptr, hWda, Aq, Bq, Aq, DPq, 0, 0);
    attn_score_kernel<<<Bq, TINq, 0, stream>>>(encat, hWda, cov, Wca, vat, in_ids, sbuf);
    softmax_ctx_kernel<<<Bq, TINq, 0, stream>>>(sbuf, enc, in_ids, tgt, tstep,
                                                cov, cl, copyp, ctx, dec_h, hc16);
    gemm_launch(stream, hc16, HCPq, W116, HCPq, b1p, z1, KZq, Bq, KZq, HCPq, 0, 1);
    z1pack_pgen_kernel<<<1, 256, 0, stream>>>(z1, z1p, ctx, dec_h, cur, emb,
                                              Wcs, bcs, Wds, Wes, pgen, sume, zsel);
    {
      int tiles = NVP / 16;                   // 3125
      int blocks = (tiles + 7) / 8;           // 391
      vocab_kernel<<<blocks, 256, 0, stream>>>(z1p, W216, b2p, tgt, tstep, sume, zsel);
    }
    finalize_step_kernel<<<1, 32, 0, stream>>>(tgt, tstep, zsel, sume, pgen, copyp, nll, cur);
  }
  final_out_kernel<<<1, 32, 0, stream>>>(nll, cl, (float*)d_out);
}
